// TimeCollapseRNN_68401649156263
// MI455X (gfx1250) — matrix operand order repaired and checked by reading, not compiled, not run on hardware
//
#include <hip/hip_runtime.h>
#include <hip/hip_bf16.h>

// ---------------------------------------------------------------------------
// Bidirectional LSTM (T=256, N=S*A=4096, C=32, H=64/dir) on gfx1250 WMMA.
// - fp16 operands / fp32 accumulate via v_wmma_f32_16x16x32_f16
// - each workgroup owns 32 batch rows for ALL timesteps (no grid sync)
// - cell state c in v8f registers, hidden state h in double-buffered LDS
//   (one s_barrier per timestep)
// - gate activations via v_tanh_f32 (TRANS op, co-executes with WMMA pipe)
// - streaming output uses non-temporal stores (keep L2 for xT + weights)
// ---------------------------------------------------------------------------

typedef _Float16 v16h __attribute__((ext_vector_type(16)));
typedef _Float16 h8   __attribute__((ext_vector_type(8)));
typedef float    v8f  __attribute__((ext_vector_type(8)));
typedef float    f4   __attribute__((ext_vector_type(4)));

#define S_DIM 32
#define C_DIM 32
#define T_DIM 256
#define A_DIM 128
#define N_DIM (S_DIM * A_DIM)   // 4096
#define H_DIM 64
#define ROWS_PER_WG 32

#if defined(__has_builtin)
#if __has_builtin(__builtin_amdgcn_tanhf)
#define USE_HW_TANH 1
#endif
#if __has_builtin(__builtin_amdgcn_rcpf)
#define USE_HW_RCP 1
#endif
#endif

__device__ __forceinline__ float fast_rcp(float x) {
#ifdef USE_HW_RCP
  return __builtin_amdgcn_rcpf(x);     // v_rcp_f32
#else
  return 1.0f / x;
#endif
}

__device__ __forceinline__ float fast_tanh(float x) {
#ifdef USE_HW_TANH
  return __builtin_amdgcn_tanhf(x);    // v_tanh_f32
#else
  return 2.0f * fast_rcp(1.0f + __expf(-2.0f * x)) - 1.0f;
#endif
}

__device__ __forceinline__ float fast_sig(float x) {
#ifdef USE_HW_TANH
  return __builtin_fmaf(__builtin_amdgcn_tanhf(0.5f * x), 0.5f, 0.5f);
#else
  return fast_rcp(1.0f + __expf(-x));
#endif
}

// x (S, C, T, A) f32  ->  xT (T, N=S*A, C) f16.  Coalesced reads along A,
// contiguous 16B stores (each thread writes 64B, threads contiguous).
// x is read exactly once -> non-temporal loads.
__global__ __launch_bounds__(128) void xpose_f16(const float* __restrict__ x,
                                                 _Float16* __restrict__ xT) {
  const int t = blockIdx.x, s = blockIdx.y, a = threadIdx.x;
  float v[C_DIM];
#pragma unroll
  for (int c = 0; c < C_DIM; ++c)
    v[c] = __builtin_nontemporal_load(
        x + ((size_t)(s * C_DIM + c) * T_DIM + t) * A_DIM + a);
  _Float16* dst = xT + ((size_t)t * N_DIM + s * A_DIM + a) * C_DIM;
#pragma unroll
  for (int g = 0; g < 4; ++g) {
    h8 p;
#pragma unroll
    for (int j = 0; j < 8; ++j) p[j] = (_Float16)v[g * 8 + j];
    *(h8*)(dst + g * 8) = p;
  }
}

__global__ __launch_bounds__(256) void bilstm_f16_wmma(
    const _Float16* __restrict__ xT,
    const float* __restrict__ Wih_f, const float* __restrict__ Whh_f,
    const float* __restrict__ bih_f, const float* __restrict__ bhh_f,
    const float* __restrict__ Wih_b, const float* __restrict__ Whh_b,
    const float* __restrict__ bih_b, const float* __restrict__ bhh_b,
    float* __restrict__ out) {
  const int dir = blockIdx.y;
  const float* Wih = dir ? Wih_b : Wih_f;
  const float* Whh = dir ? Whh_b : Whh_f;
  const float* bih = dir ? bih_b : bih_f;
  const float* bhh = dir ? bhh_b : bhh_f;

  const int row0  = blockIdx.x * ROWS_PER_WG;   // batch slab base
  const int wave  = threadIdx.x >> 5;
  const int lane  = threadIdx.x & 31;
  const int mh    = wave & 1;    // 16-row half of the slab
  const int jt    = wave >> 1;   // hidden-column tile 0..3
  const int lcol  = lane & 15;
  const int khalf = lane >> 4;

  // ---- loop-invariant B fragments (weights), f32 -> f16 in registers ----
  // B layout (KxN): lane holds column lcol; elements = 16 contiguous K,
  // lanes 0-15 -> K[0..15], lanes 16-31 -> K[16..31] of each K-block.
  v16h Bhh[4][2], Bih[4];
  float bias[4];
#pragma unroll
  for (int g = 0; g < 4; ++g) {
    const int col = g * H_DIM + jt * 16 + lcol;   // gate column 0..255
    const float* wr = Whh + col * H_DIM;          // W_hh row = B column
#pragma unroll
    for (int kt = 0; kt < 2; ++kt) {
      const float* p = wr + kt * 32 + khalf * 16; // 16 consecutive K (f32)
      f4 q0 = *(const f4*)(p);
      f4 q1 = *(const f4*)(p + 4);
      f4 q2 = *(const f4*)(p + 8);
      f4 q3 = *(const f4*)(p + 12);
      v16h b;
#pragma unroll
      for (int j = 0; j < 4; ++j) {
        b[j]      = (_Float16)q0[j];
        b[4 + j]  = (_Float16)q1[j];
        b[8 + j]  = (_Float16)q2[j];
        b[12 + j] = (_Float16)q3[j];
      }
      Bhh[g][kt] = b;
    }
    {
      const float* p = Wih + col * C_DIM + khalf * 16;  // K = C = 32
      f4 q0 = *(const f4*)(p);
      f4 q1 = *(const f4*)(p + 4);
      f4 q2 = *(const f4*)(p + 8);
      f4 q3 = *(const f4*)(p + 12);
      v16h b;
#pragma unroll
      for (int j = 0; j < 4; ++j) {
        b[j]      = (_Float16)q0[j];
        b[4 + j]  = (_Float16)q1[j];
        b[8 + j]  = (_Float16)q2[j];
        b[12 + j] = (_Float16)q3[j];
      }
      Bih[g] = b;
    }
    bias[g] = bih[col] + bhh[col];
  }

  // ---- double-buffered shared hidden state, register cell state ----
  __shared__ _Float16 hbuf[2][ROWS_PER_WG * H_DIM];   // 2 x 4 KB
  {
    h8 z;
#pragma unroll
    for (int j = 0; j < 8; ++j) z[j] = (_Float16)0.f;
    ((h8*)hbuf[0])[threadIdx.x] = z;                  // h0 = 0 in buffer 0
  }
  v8f ccell;
#pragma unroll
  for (int v = 0; v < 8; ++v) ccell[v] = 0.f;
  __syncthreads();

  // A-fragment addressing: lane holds row arow; elements e0..15 = K[kb..kb+15]
  // with kb = khalf*16 (contiguous K, matching the B fragments).
  const int arow = mh * 16 + lcol;
  const int hoff = arow * H_DIM;                 // row offset into h buffer
  const int woff = (mh * 16 + khalf * 8) * H_DIM + jt * 16 + lcol;

  // output addressing: out[s, dir*64 + j, t, a]; our slab is within one s.
  const int srow  = row0 >> 7;
  const int abase = row0 & 127;
  const int jp    = dir * H_DIM + jt * 16 + lcol;
  float* outlane = out + ((size_t)(srow * (2 * H_DIM) + jp) * T_DIM) * A_DIM
                       + abase + mh * 16 + khalf * 8;

  for (int step = 0; step < T_DIM; ++step) {
    const int tt = dir ? (T_DIM - 1 - step) : step;
    const int rb = step & 1;           // read h from this buffer
    const _Float16* hrow = hbuf[rb] + hoff;
    _Float16* hw = hbuf[rb ^ 1] + woff;

    // A fragment of x_t (16 rows x K=32) straight from global f16
    const _Float16* xr = xT + ((size_t)tt * N_DIM + row0 + arow) * C_DIM;
    h8 xlo = *(const h8*)(xr + khalf * 16);
    h8 xhi = *(const h8*)(xr + 8 + khalf * 16);
    v16h Ax = __builtin_shufflevector(xlo, xhi,
        0,1,2,3,4,5,6,7,8,9,10,11,12,13,14,15);

    if (step + 1 < T_DIM) {   // hide next step's x latency
      const int tn = dir ? (T_DIM - 2 - step) : (step + 1);
      __builtin_prefetch(xT + ((size_t)tn * N_DIM + row0 + arow) * C_DIM, 0, 1);
    }

    // A fragments of h_{t-1} (16 rows x K=64) from LDS
    h8 a0 = *(const h8*)(hrow + khalf * 16);
    h8 a1 = *(const h8*)(hrow + 8 + khalf * 16);
    h8 a2 = *(const h8*)(hrow + 32 + khalf * 16);
    h8 a3 = *(const h8*)(hrow + 40 + khalf * 16);
    v16h Ah0 = __builtin_shufflevector(a0, a1,
        0,1,2,3,4,5,6,7,8,9,10,11,12,13,14,15);
    v16h Ah1 = __builtin_shufflevector(a2, a3,
        0,1,2,3,4,5,6,7,8,9,10,11,12,13,14,15);

    // gate pre-activations: g = x@Wih.T + h@Whh.T + b  (12 WMMA / wave)
    v8f acc[4];
#pragma unroll
    for (int g = 0; g < 4; ++g) {
      v8f c;
#pragma unroll
      for (int v = 0; v < 8; ++v) c[v] = bias[g];
      c = __builtin_amdgcn_wmma_f32_16x16x32_f16(false, Ah0, false, Bhh[g][0],
                                                 (short)0, c, false, false);
      c = __builtin_amdgcn_wmma_f32_16x16x32_f16(false, Ah1, false, Bhh[g][1],
                                                 (short)0, c, false, false);
      c = __builtin_amdgcn_wmma_f32_16x16x32_f16(false, Ax,  false, Bih[g],
                                                 (short)0, c, false, false);
      acc[g] = c;
    }

    // pointwise LSTM cell update, fp32, wave-local (tanh-only, no division)
    v8f hv;
#pragma unroll
    for (int v = 0; v < 8; ++v) {
      float ig = fast_sig(acc[0][v]);
      float fg = fast_sig(acc[1][v]);
      float gg = fast_tanh(acc[2][v]);
      float og = fast_sig(acc[3][v]);
      float cc = __builtin_fmaf(fg, ccell[v], ig * gg);
      ccell[v] = cc;
      hv[v] = og * fast_tanh(cc);
    }

    // write new h into the other buffer (no race with this step's reads)
#pragma unroll
    for (int v = 0; v < 8; ++v)   // D layout: lane col jp, rows v+8*khalf
      hw[v * H_DIM] = (_Float16)hv[v];

    // out[s, jp, tt, abase + row]: rows are consecutive 'a' -> two b128
    // non-temporal stores (streaming output, never re-read)
    f4 o0, o1;
#pragma unroll
    for (int v = 0; v < 4; ++v) { o0[v] = hv[v]; o1[v] = hv[4 + v]; }
    float* po = outlane + (size_t)tt * A_DIM;
    __builtin_nontemporal_store(o0, (f4*)po);
    __builtin_nontemporal_store(o1, (f4*)(po + 4));

    __syncthreads();   // single barrier: new h visible before next step
  }
}

extern "C" void kernel_launch(void* const* d_in, const int* in_sizes, int n_in,
                              void* d_out, int out_size, void* d_ws, size_t ws_size,
                              hipStream_t stream) {
  (void)in_sizes; (void)n_in; (void)out_size; (void)ws_size;
  const float* x     = (const float*)d_in[0];
  const float* Wih_f = (const float*)d_in[1];
  const float* Whh_f = (const float*)d_in[2];
  const float* bih_f = (const float*)d_in[3];
  const float* bhh_f = (const float*)d_in[4];
  const float* Wih_b = (const float*)d_in[5];
  const float* Whh_b = (const float*)d_in[6];
  const float* bih_b = (const float*)d_in[7];
  const float* bhh_b = (const float*)d_in[8];
  float* out = (float*)d_out;

  // workspace: xT (T, N, C) f16 = 256*4096*32*2 = 64 MB
  _Float16* xT = (_Float16*)d_ws;

  xpose_f16<<<dim3(T_DIM, S_DIM), 128, 0, stream>>>(x, xT);
  bilstm_f16_wmma<<<dim3(N_DIM / ROWS_PER_WG, 2), 256, 0, stream>>>(
      xT, Wih_f, Whh_f, bih_f, bhh_f, Wih_b, Whh_b, bih_b, bhh_b, out);
}
